// HybridSO3FrameDenoiser_20117626814827
// MI455X (gfx1250) — matrix operand order repaired and checked by reading, not compiled, not run on hardware
//
#include <hip/hip_runtime.h>
#include <math.h>

typedef _Float16 f16;
typedef __attribute__((ext_vector_type(16))) _Float16 v16h;
typedef __attribute__((ext_vector_type(8)))  _Float16 v8h;
typedef __attribute__((ext_vector_type(8)))  float    v8f;

#define CDIV(a,b) (((a)+(b)-1)/(b))

// ---------------------------------------------------------------------------
// Generic WMMA GEMM: C[M,Nc] = act(A[M,K] @ B[K,Nc] (+ C if ACC))
// Row-major f32 in/out, f16 WMMA compute, f32 accumulate.
// Block: 256 threads = 8 waves. Tile: 256(M) x 64(N), K-step 32.
// Wave w computes rows [w*32, w*32+32) x 64 cols: 2 A-frags x 4 B-frags,
// 8 v_wmma_f32_16x16x32_f16 per K-step. Double-buffered LDS (1 barrier/step).
// LDS row stride = 48 halves (96B, 16B aligned) to spread banks.
// ---------------------------------------------------------------------------
#define LDST 48

template<int ACT, bool ACC>
__global__ __launch_bounds__(256) void gemm_kernel(
    const float* __restrict__ A, const float* __restrict__ B,
    float* __restrict__ C, int M, int K, int Nc)
{
  __shared__ __align__(16) f16 As[2][256 * LDST];  // [m][k]
  __shared__ __align__(16) f16 Bt[2][64 * LDST];   // [n][k] (transposed)

  const int m0   = blockIdx.x * 256;
  const int n0   = blockIdx.y * 64;
  const int tid  = threadIdx.x;
  const int wave = tid >> 5;
  const int lane = tid & 31;
  const int lm   = lane & 15;
  const int hb   = lane >> 4;

  const bool kAligned = (K & 3) == 0;
  const bool nAligned = (Nc & 3) == 0;

  v8f acc[2][4];
  #pragma unroll
  for (int mi = 0; mi < 2; mi++)
    #pragma unroll
    for (int j = 0; j < 4; j++)
      #pragma unroll
      for (int i = 0; i < 8; i++) acc[mi][j][i] = 0.f;

  // ---- staging helpers ----
  auto stageA = [&](int k0, int pb) {
    const int r  = tid;               // one row per thread (256 rows)
    const int gm = m0 + r;
    f16* dst = &As[pb][r * LDST];
    const bool fast = kAligned & (m0 + 256 <= M) & (k0 + 32 <= K); // uniform
    if (fast) {
      const float4* srow = (const float4*)(A + (size_t)gm * K + k0);
      if (k0 + 64 < K) __builtin_prefetch(A + (size_t)gm * K + k0 + 64, 0, 1);
      #pragma unroll
      for (int i = 0; i < 4; i++) {
        float4 v0 = srow[2 * i], v1 = srow[2 * i + 1];
        v8h h;
        h[0]=(f16)v0.x; h[1]=(f16)v0.y; h[2]=(f16)v0.z; h[3]=(f16)v0.w;
        h[4]=(f16)v1.x; h[5]=(f16)v1.y; h[6]=(f16)v1.z; h[7]=(f16)v1.w;
        *(v8h*)(dst + i * 8) = h;
      }
    } else {
      for (int i = 0; i < 32; i++) {
        const int gk = k0 + i;
        float v = (gm < M && gk < K) ? A[(size_t)gm * K + gk] : 0.f;
        dst[i] = (f16)v;
      }
    }
  };
  auto stageB = [&](int k0, int pb) {
    const int nb = (tid >> 4) * 4;    // 0..60
    const int kp = (tid & 15) * 2;    // 0..30
    const int gk = k0 + kp;
    const bool fast = nAligned & (n0 + 64 <= Nc) & (k0 + 32 <= K);  // uniform
    if (fast) {
      const float4 r0 = *(const float4*)(B + (size_t)gk * Nc + n0 + nb);
      const float4 r1 = *(const float4*)(B + (size_t)(gk + 1) * Nc + n0 + nb);
      f16* bp = &Bt[pb][0];
      f16* d0 = bp + (nb + 0) * LDST + kp; d0[0]=(f16)r0.x; d0[1]=(f16)r1.x;
      f16* d1 = bp + (nb + 1) * LDST + kp; d1[0]=(f16)r0.y; d1[1]=(f16)r1.y;
      f16* d2 = bp + (nb + 2) * LDST + kp; d2[0]=(f16)r0.z; d2[1]=(f16)r1.z;
      f16* d3 = bp + (nb + 3) * LDST + kp; d3[0]=(f16)r0.w; d3[1]=(f16)r1.w;
    } else {
      for (int kk = 0; kk < 2; kk++)
        for (int i = 0; i < 4; i++) {
          const int gkk = k0 + kp + kk, gn = n0 + nb + i;
          float v = (gkk < K && gn < Nc) ? B[(size_t)gkk * Nc + gn] : 0.f;
          Bt[pb][(nb + i) * LDST + kp + kk] = (f16)v;
        }
    }
  };
  auto afrag = [&](const f16* ap) {
    // 16-bit A 16x32 layout: e<8 -> k=hb*8+e ; e>=8 -> k=16+hb*8+(e-8)
    const v8h lo = *(const v8h*)(ap + hb * 8);
    const v8h hi = *(const v8h*)(ap + 16 + hb * 8);
    v16h a;
    #pragma unroll
    for (int i = 0; i < 8; i++) { a[i] = lo[i]; a[8 + i] = hi[i]; }
    return a;
  };
  auto bfrag = [&](const f16* bp) {
    // 16-bit B 32x16 layout: e<8 -> k=hb*8+e ; e>=8 -> k=16+hb*8+(e-8)
    const v8h b0 = *(const v8h*)(bp + hb * 8);
    const v8h b1 = *(const v8h*)(bp + 16 + hb * 8);
    v16h b;
    #pragma unroll
    for (int i = 0; i < 8; i++) { b[i] = b0[i]; b[8 + i] = b1[i]; }
    return b;
  };

  const int ntiles = CDIV(K, 32);
  int p = 0;
  stageA(0, 0);
  stageB(0, 0);
  for (int ti = 0; ti < ntiles; ti++) {
    __syncthreads();
    const int k0n = (ti + 1) * 32;
    if (k0n < K) { stageA(k0n, p ^ 1); stageB(k0n, p ^ 1); }
    const f16* ap0 = &As[p][(wave * 32 + lm) * LDST];
    const v16h a0 = afrag(ap0);
    const v16h a1 = afrag(ap0 + 16 * LDST);
    #pragma unroll
    for (int j = 0; j < 4; j++) {
      const v16h b = bfrag(&Bt[p][(j * 16 + lm) * LDST]);
      acc[0][j] = __builtin_amdgcn_wmma_f32_16x16x32_f16(
          false, a0, false, b, (short)0, acc[0][j], false, false);
      acc[1][j] = __builtin_amdgcn_wmma_f32_16x16x32_f16(
          false, a1, false, b, (short)0, acc[1][j], false, false);
    }
    p ^= 1;
  }

  // epilogue: C/D layout: lane(lm,hb), reg r -> (m = hb*8+r, n = lm)
  #pragma unroll
  for (int mi = 0; mi < 2; mi++) {
    #pragma unroll
    for (int j = 0; j < 4; j++) {
      const int gn = n0 + j * 16 + lm;
      if (gn >= Nc) continue;
      #pragma unroll
      for (int r = 0; r < 8; r++) {
        const int gm = m0 + wave * 32 + mi * 16 + hb * 8 + r;
        if (gm >= M) continue;
        float v = acc[mi][j][r];
        if (ACC) v += C[(size_t)gm * Nc + gn];
        if (ACT == 1) v = v > 0.f ? v : 0.f;
        else if (ACT == 2) v = 1.f / (1.f + expf(-v));
        C[(size_t)gm * Nc + gn] = v;
      }
    }
  }
}

// ---------------------------------------------------------------------------
// LayerNorm over last dim D: out = LN(x + res?) * g + b. One warp per row.
// ---------------------------------------------------------------------------
__global__ void ln_kernel(const float* __restrict__ x, const float* __restrict__ res,
                          const float* __restrict__ g, const float* __restrict__ b,
                          float* __restrict__ out, int rows, int D)
{
  const int row  = blockIdx.x * (blockDim.x >> 5) + (threadIdx.x >> 5);
  const int lane = threadIdx.x & 31;
  if (row >= rows) return;
  const float* xr = x + (size_t)row * D;
  const float* rr = res ? res + (size_t)row * D : nullptr;
  float sum = 0.f, sq = 0.f;
  for (int c = lane; c < D; c += 32) {
    float v = xr[c] + (rr ? rr[c] : 0.f);
    sum += v; sq += v * v;
  }
  for (int off = 16; off > 0; off >>= 1) {
    sum += __shfl_down(sum, off);
    sq  += __shfl_down(sq, off);
  }
  sum = __shfl(sum, 0); sq = __shfl(sq, 0);
  const float mean = sum / D;
  const float inv  = rsqrtf(sq / D - mean * mean + 1e-5f);
  float* orow = out + (size_t)row * D;
  for (int c = lane; c < D; c += 32) {
    float v = xr[c] + (rr ? rr[c] : 0.f);
    orow[c] = (v - mean) * inv * g[c] + b[c];
  }
}

// ---------------------------------------------------------------------------
__global__ void center_kernel(const float* trans, float* center, int N) {
  __shared__ float sb[3][256];
  float sx = 0, sy = 0, sz = 0;
  for (int i = threadIdx.x; i < N; i += 256) {
    sx += trans[i * 3]; sy += trans[i * 3 + 1]; sz += trans[i * 3 + 2];
  }
  sb[0][threadIdx.x] = sx; sb[1][threadIdx.x] = sy; sb[2][threadIdx.x] = sz;
  __syncthreads();
  for (int s = 128; s > 0; s >>= 1) {
    if (threadIdx.x < s)
      for (int d = 0; d < 3; d++) sb[d][threadIdx.x] += sb[d][threadIdx.x + s];
    __syncthreads();
  }
  if (threadIdx.x < 3) center[threadIdx.x] = sb[threadIdx.x][0] / N;
}

__global__ void prep_kernel(const float* trans, const float* quats, const float* center,
                            float* t, float* ts_, float* q, int N) {
  int n = blockIdx.x * blockDim.x + threadIdx.x;
  if (n >= N) return;
  for (int i = 0; i < 3; i++) {
    float v = trans[n * 3 + i] - center[i];
    t[n * 3 + i] = v; ts_[n * 3 + i] = v * 0.1f;
  }
  float w = quats[n*4], x = quats[n*4+1], y = quats[n*4+2], z = quats[n*4+3];
  float inv = rsqrtf(w*w + x*x + y*y + z*z);
  q[n*4] = w*inv; q[n*4+1] = x*inv; q[n*4+2] = y*inv; q[n*4+3] = z*inv;
}

__global__ void geo_feats_kernel(const float* t, const int* ei, int E, float* z) {
  int e = blockIdx.x * blockDim.x + threadIdx.x;
  if (e >= E) return;
  int si = ei[e], di = ei[E + e];
  float dx = t[di*3]-t[si*3], dy = t[di*3+1]-t[si*3+1], dz = t[di*3+2]-t[si*3+2];
  float dist = sqrtf(dx*dx + dy*dy + dz*dz);
  float* zr = z + (size_t)e * 128;
  const float inv_sigma = 64.f / 20.f;
  for (int i = 0; i < 64; i++) {
    float mu = 20.f * i / 63.f;
    float u = (dist - mu) * inv_sigma;
    zr[i] = expf(-u * u);
  }
  float off = (float)(di - si);
  const float L = 9.210340371976184f; // ln(1e4)
  for (int j = 0; j < 32; j++) {
    float ang = off * expf(-L * j / 32.f);
    zr[64 + j] = sinf(ang);
    zr[96 + j] = cosf(ang);
  }
}

__global__ void node_input_kernel(const float* tsv, const float* nm, float* ni, int N) {
  int n = blockIdx.x * blockDim.x + threadIdx.x;
  if (n >= N) return;
  float* r = ni + (size_t)n * 321;
  const float L = 9.210340371976184f;
  float fn = (float)n;
  for (int j = 0; j < 128; j++) {
    float ang = fn * expf(-L * j / 128.f);
    r[j] = sinf(ang); r[128 + j] = cosf(ang);
  }
  float tv = tsv[n];
  for (int j = 0; j < 64; j++) {
    float u = (tv - (float)j / 63.f) * 64.f;
    r[256 + j] = expf(-u * u);
  }
  r[320] = nm[n];
}

__global__ void deg_kernel(const int* ei, int E, float* deg) {
  int e = blockIdx.x * blockDim.x + threadIdx.x;
  if (e >= E) return;
  atomicAdd(&deg[ei[E + e]], 1.f);
}

__global__ void quat2rot_kernel(const float* q, float* R, int N) {
  int n = blockIdx.x * blockDim.x + threadIdx.x;
  if (n >= N) return;
  float w = q[n*4], x = q[n*4+1], y = q[n*4+2], z = q[n*4+3];
  float inv = rsqrtf(w*w + x*x + y*y + z*z);
  w *= inv; x *= inv; y *= inv; z *= inv;
  float* Rn = R + (size_t)n * 9;
  Rn[0] = 1-2*(y*y+z*z); Rn[1] = 2*(x*y-w*z); Rn[2] = 2*(x*z+w*y);
  Rn[3] = 2*(x*y+w*z);   Rn[4] = 1-2*(x*x+z*z); Rn[5] = 2*(y*z-w*x);
  Rn[6] = 2*(x*z-w*y);   Rn[7] = 2*(y*z+w*x); Rn[8] = 1-2*(x*x+y*y);
}

__global__ void pts_glob_kernel(float* buf, const float* R, const float* ts, int N, int npts) {
  int idx = blockIdx.x * blockDim.x + threadIdx.x;
  if (idx >= N * npts) return;
  int n = idx / npts, p = idx - n * npts;
  float* v = buf + (size_t)n * npts * 3 + p * 3;
  const float* Rn = R + (size_t)n * 9;
  float x = v[0], y = v[1], z = v[2];
  v[0] = Rn[0]*x + Rn[1]*y + Rn[2]*z + ts[n*3];
  v[1] = Rn[3]*x + Rn[4]*y + Rn[5]*z + ts[n*3+1];
  v[2] = Rn[6]*x + Rn[7]*y + Rn[8]*z + ts[n*3+2];
}

__device__ __forceinline__ void atomicMaxFloat(float* addr, float value) {
  if (value >= 0.f) atomicMax((int*)addr, __float_as_int(value));
  else              atomicMin((unsigned int*)addr, __float_as_uint(value));
}

__global__ void fill_kernel(float* p, float v, size_t n) {
  size_t i = (size_t)blockIdx.x * blockDim.x + threadIdx.x;
  if (i < n) p[i] = v;
}

__global__ void edge_logits_kernel(const float* qb, const float* kb, const float* qp,
                                   const float* kp, const float* be, const float* gamma,
                                   const int* ei, int E, float* att, float* m8) {
  int idx = blockIdx.x * blockDim.x + threadIdx.x;
  if (idx >= E * 8) return;
  int e = idx >> 3, h = idx & 7;
  int si = ei[e], di = ei[E + e];
  const float* qv = qb + (size_t)di * 128 + h * 16;
  const float* kv = kb + (size_t)si * 128 + h * 16;
  float sc = 0.f;
  for (int i = 0; i < 16; i++) sc += qv[i] * kv[i];
  sc *= 0.25f;  // 1/sqrt(16)
  const float* qpp = qp + (size_t)di * 96 + h * 12;
  const float* kpp = kp + (size_t)si * 96 + h * 12;
  float d2 = 0.f;
  for (int i = 0; i < 12; i++) { float df = qpp[i] - kpp[i]; d2 += df * df; }
  float sp = log1pf(expf(gamma[h]));
  const float wc = 0.23570226039551584f;  // sqrt(2/36)
  float logit = (sc + be[(size_t)e * 8 + h] - 0.5f * sp * wc * d2) * 0.57735026918962576f;
  att[(size_t)e * 8 + h] = logit;
  atomicMaxFloat(&m8[di * 8 + h], logit);
}

__global__ void finite_fix_kernel(float* m8, int n) {
  int i = blockIdx.x * blockDim.x + threadIdx.x;
  if (i < n && m8[i] < -5e29f) m8[i] = 0.f;
}

__global__ void exp_den_kernel(float* att, const float* m8, float* den, const int* ei, int E) {
  int idx = blockIdx.x * blockDim.x + threadIdx.x;
  if (idx >= E * 8) return;
  int e = idx >> 3, h = idx & 7;
  int di = ei[E + e];
  float v = expf(att[idx] - m8[di * 8 + h]);
  att[idx] = v;
  atomicAdd(&den[di * 8 + h], v);
}

__global__ void attn_norm_kernel(float* att, const float* den, const int* ei, int E) {
  int idx = blockIdx.x * blockDim.x + threadIdx.x;
  if (idx >= E * 8) return;
  int e = idx >> 3, h = idx & 7;
  int di = ei[E + e];
  att[idx] = att[idx] / (den[di * 8 + h] + 1e-9f);
}

__global__ void agg_v_kernel(const float* att, const float* vb, const float* vp,
                             const int* ei, int E, float* o, float* opt) {
  int idx = blockIdx.x * blockDim.x + threadIdx.x;
  if (idx >= E * 8) return;
  int e = idx >> 3, h = idx & 7;
  int si = ei[e], di = ei[E + e];
  float a = att[(size_t)e * 8 + h];
  const float* vs = vb + (size_t)si * 128 + h * 16;
  float* od = o + (size_t)di * 128 + h * 16;
  for (int i = 0; i < 16; i++) atomicAdd(&od[i], a * vs[i]);
  const float* ps = vp + (size_t)si * 192 + h * 24;
  float* pd = opt + (size_t)di * 192 + h * 24;
  for (int i = 0; i < 24; i++) atomicAdd(&pd[i], a * ps[i]);
}

__global__ void agg_z_kernel(const float* att, const float* z, const int* ei, int E, float* oz) {
  int idx = blockIdx.x * blockDim.x + threadIdx.x;
  if (idx >= E * 128) return;
  int e = idx >> 7, c = idx & 127;
  int di = ei[E + e];
  float zv = z[(size_t)e * 128 + c];
  for (int h = 0; h < 8; h++)
    atomicAdd(&oz[(size_t)di * 1024 + h * 128 + c], att[(size_t)e * 8 + h] * zv);
}

__global__ void build_cat_kernel(const float* o, const float* oz, const float* opt,
                                 const float* R, const float* ts, float* cat, int N) {
  int n = blockIdx.x * blockDim.x + threadIdx.x;
  if (n >= N) return;
  float* c = cat + (size_t)n * 1408;
  const float* on = o + (size_t)n * 128;
  for (int i = 0; i < 128; i++) c[i] = on[i];
  const float* ozn = oz + (size_t)n * 1024;
  for (int i = 0; i < 1024; i++) c[128 + i] = ozn[i];
  const float* Rn = R + (size_t)n * 9;
  float tx = ts[n*3], ty = ts[n*3+1], tz = ts[n*3+2];
  for (int hp = 0; hp < 64; hp++) {
    const float* pn = opt + (size_t)n * 192 + hp * 3;
    float wx = pn[0] - tx, wy = pn[1] - ty, wz = pn[2] - tz;
    float lx = Rn[0]*wx + Rn[3]*wy + Rn[6]*wz;  // R^T
    float ly = Rn[1]*wx + Rn[4]*wy + Rn[7]*wz;
    float lz = Rn[2]*wx + Rn[5]*wy + Rn[8]*wz;
    c[1152 + hp*3] = lx; c[1152 + hp*3+1] = ly; c[1152 + hp*3+2] = lz;
    c[1344 + hp] = sqrtf(lx*lx + ly*ly + lz*lz + 1e-8f);
  }
}

__global__ void so3_msg_kernel(const float* emb, const float* gate, const int* ei,
                               int E, float* msg) {
  int idx = blockIdx.x * blockDim.x + threadIdx.x;
  if (idx >= E * 64) return;
  int e = idx >> 6, c = idx & 63;
  int si = ei[e], di = ei[E + e];
  atomicAdd(&msg[(size_t)di * 64 + c], emb[(size_t)si * 64 + c] * gate[(size_t)e * 64 + c]);
}

__global__ void so3_update_kernel(float* u, const float* emb, const float* msg,
                                  const float* deg, int N) {
  int idx = blockIdx.x * blockDim.x + threadIdx.x;
  if (idx >= N * 64) return;
  int n = idx >> 6;
  u[idx] = emb[idx] + msg[idx] / (deg[n] + 1.f);
}

__global__ void bb_kernel(float* q, float* ts, const float* R, const float* g6,
                          const float* nm, int N) {
  int n = blockIdx.x * blockDim.x + threadIdx.x;
  if (n >= N) return;
  float m = nm[n], s2 = m * m;
  float u[6];
  for (int i = 0; i < 6; i++) u[i] = g6[(size_t)n * 6 + i] * s2;
  const float* Rn = R + (size_t)n * 9;
  for (int i = 0; i < 3; i++)
    ts[n*3+i] += Rn[i*3]*u[3] + Rn[i*3+1]*u[4] + Rn[i*3+2]*u[5];
  float inv = rsqrtf(1.f + u[0]*u[0] + u[1]*u[1] + u[2]*u[2]);
  float bw = inv, bx = u[0]*inv, by = u[1]*inv, bz = u[2]*inv;
  float aw = q[n*4], ax = q[n*4+1], ay = q[n*4+2], az = q[n*4+3];
  q[n*4]   = aw*bw - ax*bx - ay*by - az*bz;
  q[n*4+1] = aw*bx + ax*bw + ay*bz - az*by;
  q[n*4+2] = aw*by - ax*bz + ay*bw + az*bx;
  q[n*4+3] = aw*bz + ax*by - ay*bx + az*bw;
}

__global__ void et_gather_kernel(float* pre, const float* t1, const float* t2,
                                 const int* ei, int E) {
  int idx = blockIdx.x * blockDim.x + threadIdx.x;
  if (idx >= E * 128) return;
  int e = idx >> 7, c = idx & 127;
  int si = ei[e], di = ei[E + e];
  float v = pre[(size_t)e*128+c] + t1[(size_t)si*128+c] + t2[(size_t)di*128+c];
  pre[(size_t)e*128+c] = v > 0.f ? v : 0.f;
}

__global__ void output_kernel(const float* s, const float* ts, const float* center,
                              const float* pbuf, float* out, int N) {
  int n = blockIdx.x * blockDim.x + threadIdx.x;
  if (n >= N) return;
  float* o = out + (size_t)n * 261;
  const float* sr = s + (size_t)n * 256;
  for (int c = 0; c < 256; c++) o[c] = sr[c];
  for (int i = 0; i < 3; i++) o[256 + i] = ts[n*3+i] * 10.f + center[i];
  float p0 = pbuf[n*2], p1 = pbuf[n*2+1];
  float inv = rsqrtf(p0*p0 + p1*p1 + 1e-8f);
  o[259] = p0 * inv; o[260] = p1 * inv;
}

// ---------------------------------------------------------------------------
// Host side
// ---------------------------------------------------------------------------
struct EtOff  { size_t down, ln_g, ln_b, w1, w2; };
struct IpaOff { size_t gamma, wb, wk, wkp, wo, wq, wqp, wv, wvp; };
struct LayerOff {
  EtOff et, ets; IpaOff sp, sq;
  size_t ln1g, ln1b, ln2g, ln2b, ln3g, ln3b;
  size_t nt0, nt1, nt2, ntg, ntb;
  size_t wbb, wc, we, wg;
};

static void gemm(hipStream_t st, const float* A, const float* B, float* C,
                 int M, int K, int Nc, int act = 0, bool acc = false) {
  dim3 g(CDIV(M, 256), CDIV(Nc, 64)), b(256);
  if (acc)            gemm_kernel<0, true ><<<g, b, 0, st>>>(A, B, C, M, K, Nc);
  else if (act == 1)  gemm_kernel<1, false><<<g, b, 0, st>>>(A, B, C, M, K, Nc);
  else if (act == 2)  gemm_kernel<2, false><<<g, b, 0, st>>>(A, B, C, M, K, Nc);
  else                gemm_kernel<0, false><<<g, b, 0, st>>>(A, B, C, M, K, Nc);
}

static void ln(hipStream_t st, const float* x, const float* res, const float* g,
               const float* b, float* out, int rows, int D) {
  ln_kernel<<<CDIV(rows, 8), 256, 0, st>>>(x, res, g, b, out, rows, D);
}

extern "C" void kernel_launch(void* const* d_in, const int* in_sizes, int n_in,
                              void* d_out, int out_size, void* d_ws, size_t ws_size,
                              hipStream_t stream) {
  const float* trans = (const float*)d_in[0];
  const float* quats = (const float*)d_in[1];
  const float* tsin  = (const float*)d_in[2];
  const float* nm    = (const float*)d_in[3];
  const float* P     = (const float*)d_in[4];
  const int*   ei    = (const int*)d_in[6];
  const int*   eis   = (const int*)d_in[7];
  const int N  = in_sizes[0] / 3;
  const int E  = in_sizes[6] / 2;
  const int ES = in_sizes[7] / 2;

  // -------- param offsets (jax tree_flatten: dict keys sorted) --------
  size_t cur = 0;
  auto take = [&](size_t n) { size_t o = cur; cur += n; return o; };
  const size_t ee0 = take(16384), ee1 = take(16384), ee2 = take(16384);
  const size_t eelng = take(128), eelnb = take(128);
  const size_t en0 = take(321 * 256), en1 = take(65536), en2 = take(65536);
  const size_t enlng = take(256), enlnb = take(256);
  LayerOff L[4];
  for (int l = 0; l < 4; l++) {
    auto et_off = [&]() { EtOff e; e.down = take(256*128); e.ln_g = take(128);
      e.ln_b = take(128); e.w1 = take(384*128); e.w2 = take(128*128); return e; };
    auto ipa_off = [&]() { IpaOff p; p.gamma = take(8); p.wb = take(128*8);
      p.wk = take(256*128); p.wkp = take(256*96); p.wo = take(1408*256);
      p.wq = take(256*128); p.wqp = take(256*96); p.wv = take(256*128);
      p.wvp = take(256*192); return p; };
    L[l].et = et_off(); L[l].ets = et_off();
    L[l].sp = ipa_off(); L[l].sq = ipa_off();
    L[l].ln1g = take(256); L[l].ln1b = take(256);
    L[l].ln2g = take(256); L[l].ln2b = take(256);
    L[l].ln3g = take(256); L[l].ln3b = take(256);
    L[l].nt0 = take(65536); L[l].nt1 = take(65536); L[l].nt2 = take(65536);
    L[l].ntg = take(256); L[l].ntb = take(256);
    L[l].wbb = take(256*6); L[l].wc = take(320*64);
    L[l].we = take(64*256); L[l].wg = take(128*64);
  }
  const size_t psi_off = take(512);

  // -------- workspace carve --------
  char* wsb = (char*)d_ws;
  size_t woff = 0;
  auto alloc = [&](size_t nfl) {
    float* p = (float*)(wsb + woff);
    woff += ((nfl * 4 + 255) / 256) * 256;
    return p;
  };
  float* center = alloc(4);
  float* t    = alloc((size_t)N*3);   float* ts_ = alloc((size_t)N*3);
  float* qv   = alloc((size_t)N*4);   float* R   = alloc((size_t)N*9);
  float* uvec6= alloc((size_t)N*6);
  float* s    = alloc((size_t)N*256); float* sA  = alloc((size_t)N*256);
  float* sB   = alloc((size_t)N*256); float* delta = alloc((size_t)N*256);
  float* ni   = alloc((size_t)N*321);
  float* u    = alloc((size_t)N*64);  float* emb = alloc((size_t)N*64);
  float* msg  = alloc((size_t)N*64);  float* deg = alloc((size_t)N);
  float* qb   = alloc((size_t)N*128); float* kb  = alloc((size_t)N*128);
  float* vb   = alloc((size_t)N*128);
  float* qp   = alloc((size_t)N*96);  float* kp  = alloc((size_t)N*96);
  float* vp   = alloc((size_t)N*192);
  float* o    = alloc((size_t)N*128); float* opt = alloc((size_t)N*192);
  float* oz   = alloc((size_t)N*1024);float* cat = alloc((size_t)N*1408);
  float* hn   = alloc((size_t)N*128); float* t1  = alloc((size_t)N*128);
  float* t2   = alloc((size_t)N*128);
  float* m8   = alloc((size_t)N*8);   float* den8 = alloc((size_t)N*8);
  float* pbuf = alloc((size_t)N*2);
  float* z    = alloc((size_t)E*128); float* zs  = alloc((size_t)ES*128);
  float* escr = alloc((size_t)E*128);
  float* be   = alloc((size_t)E*8);   float* att = alloc((size_t)E*8);
  (void)ws_size; (void)n_in; (void)out_size;

  // -------- prologue --------
  center_kernel<<<1, 256, 0, stream>>>(trans, center, N);
  prep_kernel<<<CDIV(N,256), 256, 0, stream>>>(trans, quats, center, t, ts_, qv, N);
  hipMemsetAsync(u, 0, (size_t)N*64*4, stream);
  hipMemsetAsync(deg, 0, (size_t)N*4, stream);
  deg_kernel<<<CDIV(E,256), 256, 0, stream>>>(ei, E, deg);
  geo_feats_kernel<<<CDIV(E,256), 256, 0, stream>>>(t, ei, E, z);
  geo_feats_kernel<<<CDIV(ES,256), 256, 0, stream>>>(t, eis, ES, zs);
  node_input_kernel<<<CDIV(N,256), 256, 0, stream>>>(tsin, nm, ni, N);

  // s = mlp3(ni, embed_node)
  gemm(stream, ni, P+en0, sA, N, 321, 256, 1);
  gemm(stream, sA, P+en1, sB, N, 256, 256, 1);
  gemm(stream, sB, P+en2, sA, N, 256, 256, 0);
  ln(stream, sA, nullptr, P+enlng, P+enlnb, s, N, 256);
  // z = mlp3(z, embed_edge)   (zs stays raw, per reference)
  gemm(stream, z, P+ee0, escr, E, 128, 128, 1);
  gemm(stream, escr, P+ee1, z, E, 128, 128, 1);
  gemm(stream, z, P+ee2, escr, E, 128, 128, 0);
  ln(stream, escr, nullptr, P+eelng, P+eelnb, z, E, 128);

  auto run_ipa = [&](const float* zbuf, const int* eic, int Ec, const IpaOff& po) {
    gemm(stream, s, P+po.wq, qb, N, 256, 128);
    gemm(stream, s, P+po.wk, kb, N, 256, 128);
    gemm(stream, s, P+po.wv, vb, N, 256, 128);
    gemm(stream, s, P+po.wqp, qp, N, 256, 96);
    gemm(stream, s, P+po.wkp, kp, N, 256, 96);
    gemm(stream, s, P+po.wvp, vp, N, 256, 192);
    pts_glob_kernel<<<CDIV(N*32,256), 256, 0, stream>>>(qp, R, ts_, N, 32);
    pts_glob_kernel<<<CDIV(N*32,256), 256, 0, stream>>>(kp, R, ts_, N, 32);
    pts_glob_kernel<<<CDIV(N*64,256), 256, 0, stream>>>(vp, R, ts_, N, 64);
    gemm(stream, zbuf, P+po.wb, be, Ec, 128, 8);
    fill_kernel<<<CDIV((size_t)N*8,256), 256, 0, stream>>>(m8, -1e30f, (size_t)N*8);
    edge_logits_kernel<<<CDIV(Ec*8,256), 256, 0, stream>>>(
        qb, kb, qp, kp, be, P+po.gamma, eic, Ec, att, m8);
    finite_fix_kernel<<<CDIV(N*8,256), 256, 0, stream>>>(m8, N*8);
    hipMemsetAsync(den8, 0, (size_t)N*8*4, stream);
    exp_den_kernel<<<CDIV(Ec*8,256), 256, 0, stream>>>(att, m8, den8, eic, Ec);
    attn_norm_kernel<<<CDIV(Ec*8,256), 256, 0, stream>>>(att, den8, eic, Ec);
    hipMemsetAsync(o, 0, (size_t)N*128*4, stream);
    hipMemsetAsync(opt, 0, (size_t)N*192*4, stream);
    hipMemsetAsync(oz, 0, (size_t)N*1024*4, stream);
    agg_v_kernel<<<CDIV(Ec*8,256), 256, 0, stream>>>(att, vb, vp, eic, Ec, o, opt);
    agg_z_kernel<<<CDIV(Ec*128,256), 256, 0, stream>>>(att, zbuf, eic, Ec, oz);
    build_cat_kernel<<<CDIV(N,256), 256, 0, stream>>>(o, oz, opt, R, ts_, cat, N);
    gemm(stream, cat, P+po.wo, delta, N, 1408, 256);
  };

  for (int l = 0; l < 4; l++) {
    quat2rot_kernel<<<CDIV(N,256), 256, 0, stream>>>(qv, R, N);
    run_ipa(z, ei, E, L[l].sp);
    ln(stream, s, delta, P+L[l].ln1g, P+L[l].ln1b, s, N, 256);
    run_ipa(zs, eis, ES, L[l].sq);
    ln(stream, s, delta, P+L[l].ln2g, P+L[l].ln2b, s, N, 256);
    // SO3 scalar branch: emb = [u,s] @ wc
    gemm(stream, u, P+L[l].wc, emb, N, 64, 64, 0, false);
    gemm(stream, s, P+L[l].wc + 64*64, emb, N, 256, 64, 0, true);
    gemm(stream, z, P+L[l].wg, escr, E, 128, 64, 2);  // gate = sigmoid(z@wg)
    hipMemsetAsync(msg, 0, (size_t)N*64*4, stream);
    so3_msg_kernel<<<CDIV(E*64,256), 256, 0, stream>>>(emb, escr, ei, E, msg);
    so3_update_kernel<<<CDIV(N*64,256), 256, 0, stream>>>(u, emb, msg, deg, N);
    gemm(stream, u, P+L[l].we, delta, N, 64, 256);
    ln(stream, s, delta, P+L[l].ln3g, P+L[l].ln3b, s, N, 256);
    // node transition MLP
    gemm(stream, s, P+L[l].nt0, sA, N, 256, 256, 1);
    gemm(stream, sA, P+L[l].nt1, sB, N, 256, 256, 1);
    gemm(stream, sB, P+L[l].nt2, delta, N, 256, 256, 0);
    ln(stream, s, delta, P+L[l].ntg, P+L[l].ntb, s, N, 256);
    // backbone update
    gemm(stream, s, P+L[l].wbb, uvec6, N, 256, 6);
    bb_kernel<<<CDIV(N,256), 256, 0, stream>>>(qv, ts_, R, uvec6, nm, N);
    // edge transition (z): pre = z@w1_lo + (hn@w1_mid)[src] + (hn@w1_hi)[dst]
    gemm(stream, s, P+L[l].et.down, hn, N, 256, 128);
    gemm(stream, hn, P+L[l].et.w1 + 128*128, t1, N, 128, 128);
    gemm(stream, hn, P+L[l].et.w1 + 256*128, t2, N, 128, 128);
    gemm(stream, z, P+L[l].et.w1, escr, E, 128, 128);
    et_gather_kernel<<<CDIV(E*128,256), 256, 0, stream>>>(escr, t1, t2, ei, E);
    gemm(stream, escr, P+L[l].et.w2, z, E, 128, 128);
    ln(stream, z, nullptr, P+L[l].et.ln_g, P+L[l].et.ln_b, z, E, 128);
    // edge transition (zs)
    gemm(stream, s, P+L[l].ets.down, hn, N, 256, 128);
    gemm(stream, hn, P+L[l].ets.w1 + 128*128, t1, N, 128, 128);
    gemm(stream, hn, P+L[l].ets.w1 + 256*128, t2, N, 128, 128);
    gemm(stream, zs, P+L[l].ets.w1, escr, ES, 128, 128);
    et_gather_kernel<<<CDIV(ES*128,256), 256, 0, stream>>>(escr, t1, t2, eis, ES);
    gemm(stream, escr, P+L[l].ets.w2, zs, ES, 128, 128);
    ln(stream, zs, nullptr, P+L[l].ets.ln_g, P+L[l].ets.ln_b, zs, ES, 128);
  }

  gemm(stream, s, P+psi_off, pbuf, N, 256, 2);
  output_kernel<<<CDIV(N,256), 256, 0, stream>>>(s, ts_, center, pbuf,
                                                 (float*)d_out, N);
}
